// FoldTemporalConvTranspose1d_19490561590069
// MI455X (gfx1250) — matrix operand order repaired and checked by reading, not compiled, not run on hardware
//
#include <hip/hip_runtime.h>

// Problem constants (from the reference)
#define B_SZ   8
#define CIN    512
#define COUT   512
#define LLEN   2048
#define NTAP   8
#define TAPG   4      // taps staged per group (2 groups)

// Tiling
#define BM     128    // co rows per block (2 wave-rows of 64)
#define BN     256    // l cols per block (4 wave-cols of 64)
#define KC     32     // contraction chunk = WMMA K
#define XCOLS  (BN + NTAP)          // 264: BN cols + 8-col left halo
#define XSTRW  20                   // uint words per X column (40 bf16 = 80B, 16B-aligned)
#define WSTRW  20                   // uint words per W row   (40 bf16 = 80B, 16B-aligned)

typedef __attribute__((ext_vector_type(16))) __bf16 v16bf;
typedef __attribute__((ext_vector_type(8)))  float  v8f;

__device__ __forceinline__ unsigned pack_bf16(float a, float b) {
    union { __bf16 h[2]; unsigned u; } r;
    r.h[0] = (__bf16)a;
    r.h[1] = (__bf16)b;
    return r.u;
}

__global__ __launch_bounds__(256)
void fold_conv_wmma_kernel(const float* __restrict__ x,
                           const float* __restrict__ W,
                           const float* __restrict__ bias,
                           float* __restrict__ y) {
    // LDS: W tiles for 4 taps [tap][row][c], X tile col-major [col][c]
    __shared__ unsigned sW32[TAPG * BM * WSTRW];  // 4*128*20*4 = 40960 B
    __shared__ unsigned sX32[XCOLS * XSTRW];      // 264*20*4   = 21120 B
    const unsigned short* sW16 = (const unsigned short*)sW32;  // 40 bf16/row view
    const unsigned short* sX16 = (const unsigned short*)sX32;  // 40 bf16/col view

    const int l0  = blockIdx.x * BN;
    const int co0 = blockIdx.y * BM;
    const int b   = blockIdx.z;

    const int tid    = threadIdx.x;
    const int wave   = tid >> 5;
    const int lane   = tid & 31;
    const int lane16 = lane & 15;
    const int laneHi = lane >> 4;
    const int waveM  = wave >> 2;    // 0..1 : 64 co rows each
    const int waveN  = wave & 3;     // 0..3 : 64 l cols each

    // 16 accumulators per wave: 4 M-tiles x 4 N-tiles (64x64 per wave)
    v8f acc[4][4];
    #pragma unroll
    for (int mt = 0; mt < 4; ++mt)
        #pragma unroll
        for (int nt = 0; nt < 4; ++nt)
            acc[mt][nt] = (v8f){0.f,0.f,0.f,0.f,0.f,0.f,0.f,0.f};

    const float* xb = x + (size_t)b * CIN * LLEN;

    for (int cc = 0; cc < CIN; cc += KC) {
        // ---- stage X once per c-chunk: 264 cols x 16 c-pairs, col-major ----
        for (int i = tid; i < XCOLS * (KC / 2); i += 256) {
            int col = i % XCOLS;                   // consecutive tid -> consecutive l (coalesced)
            int cp  = i / XCOLS;                   // c = 2*cp
            int gl  = l0 - (NTAP - 1) + col;       // global l with left halo
            float a = 0.f, c1 = 0.f;
            if (gl >= 0 && gl < LLEN) {
                const float* xc = xb + (size_t)(cc + 2 * cp) * LLEN + gl;
                a  = xc[0];
                c1 = xc[LLEN];
            }
            sX32[col * XSTRW + cp] = pack_bf16(a, c1);
        }

        #pragma unroll
        for (int g = 0; g < NTAP / TAPG; ++g) {
            // ---- stage W group g: 4 taps x 128 rows x 16 c-pairs ----
            #pragma unroll 4
            for (int i = tid; i < TAPG * BM * (KC / 2); i += 256) {
                int cp = i & 15;                   // c = 2*cp
                int r  = (i >> 4) & (BM - 1);
                int k  = i >> 11;                  // i / (BM*KC/2), 0..3
                const float2 wv = *(const float2*)
                    &W[(size_t)((g * TAPG + k) * COUT + co0 + r) * CIN + (cc + 2 * cp)];
                sW32[(k * BM + r) * WSTRW + cp] = pack_bf16(wv.x, wv.y);
            }
            __syncthreads();

            // ---- compute: 4 taps x 16 WMMA, 16 b128 LDS loads per tap ----
            #pragma unroll
            for (int k = 0; k < TAPG; ++k) {
                const int kk = g * TAPG + k;       // global tap (column shift)
                // B fragments (32x16 bf16): lane = column, 16 contiguous K per lane
                v16bf bf[4];
                #pragma unroll
                for (int nt = 0; nt < 4; ++nt) {
                    int col = waveN * 64 + nt * 16 + lane16 + (NTAP - 1) - kk;
                    const uint4* p = (const uint4*)&sX16[col * (2 * XSTRW) + laneHi * 16];
                    ((uint4*)&bf[nt])[0] = p[0];
                    ((uint4*)&bf[nt])[1] = p[1];
                }
                #pragma unroll
                for (int mt = 0; mt < 4; ++mt) {
                    // A frag (16x32 bf16): lanes 0-15 K{0..15}, lanes 16-31 K{16..31}
                    int row = waveM * 64 + mt * 16 + lane16;
                    const unsigned short* wrow = &sW16[(k * BM + row) * (2 * WSTRW)];
                    v16bf af;
                    ((uint4*)&af)[0] = *(const uint4*)&wrow[laneHi * 16];
                    ((uint4*)&af)[1] = *(const uint4*)&wrow[laneHi * 16 + 8];
                    #pragma unroll
                    for (int nt = 0; nt < 4; ++nt) {
                        acc[mt][nt] = __builtin_amdgcn_wmma_f32_16x16x32_bf16(
                            false, af, false, bf[nt],
                            (short)0, acc[mt][nt], false, false);
                    }
                }
            }
            __syncthreads();
        }
    }

    // ---- epilogue: C/D layout => M = j + 8*laneHi, N = lane16 ----
    #pragma unroll
    for (int mt = 0; mt < 4; ++mt) {
        #pragma unroll
        for (int j = 0; j < 8; ++j) {
            int co = co0 + waveM * 64 + mt * 16 + laneHi * 8 + j;
            float bv = bias[co];
            #pragma unroll
            for (int nt = 0; nt < 4; ++nt) {
                int l = l0 + waveN * 64 + nt * 16 + lane16;
                y[((size_t)b * COUT + co) * (size_t)LLEN + l] = acc[mt][nt][j] + bv;
            }
        }
    }
}

extern "C" void kernel_launch(void* const* d_in, const int* in_sizes, int n_in,
                              void* d_out, int out_size, void* d_ws, size_t ws_size,
                              hipStream_t stream) {
    const float* x  = (const float*)d_in[0];   // (8, 512, 2048) fp32
    const float* W  = (const float*)d_in[1];   // (4096, 512)    fp32
    const float* bi = (const float*)d_in[2];   // (512,)         fp32
    float* y = (float*)d_out;                  // (8, 512, 2048) fp32

    dim3 grid(LLEN / BN, COUT / BM, B_SZ);     // (8, 4, 8)
    fold_conv_wmma_kernel<<<grid, 256, 0, stream>>>(x, W, bi, y);
}
